// MeshDownMP_62947040690378
// MI455X (gfx1250) — matrix operand order repaired and checked by reading, not compiled, not run on hardware
//
#include <hip/hip_runtime.h>
#include <hip/hip_bf16.h>

// ---------- types for WMMA ----------
typedef __attribute__((ext_vector_type(16))) __bf16 v16bf;
typedef __attribute__((ext_vector_type(8)))  float  v8f;
typedef __attribute__((ext_vector_type(4)))  float  f32x4;   // clang-native vec for NT store

union BF16x16 { uint4 q[2]; v16bf m; };

__device__ __forceinline__ unsigned short f2bf(float f) {
    unsigned u = __float_as_uint(f);
    u += 0x7fffu + ((u >> 16) & 1u);      // round-to-nearest-even
    return (unsigned short)(u >> 16);
}

// Branchless SELU: no EXEC-mask divergence between WMMA chains.
__device__ __forceinline__ float selu_f(float x) {
    const float alpha = 1.6732632423543772f;
    const float scale = 1.0507009873554805f;
    float em = __expf(fminf(x, 0.f)) - 1.f;       // exp path always executed
    return scale * (x > 0.f ? x : alpha * em);    // v_cndmask, not a branch
}

// A fragment (16x32 bf16, MxK) from an LDS tile with `stride` elements per row.
// Lane<16: M=lane, K = k0+[0..15]
// Lane>=16: M=lane-16, K = k0+16+[0..15]
__device__ __forceinline__ v16bf load_a_frag(const unsigned short* base,
                                             int lane, int k0, int stride) {
    int M    = lane & 15;
    int koff = (lane & 16) ? 16 : 0;
    const unsigned short* p = base + M * stride + k0 + koff;
    BF16x16 u;
    u.q[0] = *(const uint4*)(p);
    u.q[1] = *(const uint4*)(p + 8);
    return u.m;
}

// B fragment from pre-packed weights: 32 lanes x 16 bf16 contiguous per fragment.
__device__ __forceinline__ v16bf load_b_frag(const unsigned short* pack,
                                             int frag, int lane) {
    const uint4* p = (const uint4*)(pack + (((frag << 5) + lane) << 4));
    BF16x16 u;
    u.q[0] = p[0];
    u.q[1] = p[1];
    return u.m;
}

// ---------- prep kernels ----------
__global__ void zero_f4_kernel(float4* __restrict__ p, int n4) {
    int g = blockIdx.x * blockDim.x + threadIdx.x;
    if (g < n4) p[g] = make_float4(0.f, 0.f, 0.f, 0.f);
}

// Repack W0[256x128] and W1[128x128] (row-major f32, [K][N]) into bf16
// B-fragment order: frag=(n_tile,k_tile), lane, 16 K-values contiguous.
// Lane<16: N=n_tile*16+lane,   K=k_tile*32 + j    (j=0..15)
// Lane>=16: N=n_tile*16+lane-16, K=k_tile*32+16+j
__global__ void pack_weights_kernel(const float* __restrict__ W0,
                                    const float* __restrict__ W1,
                                    unsigned short* __restrict__ w0p,
                                    unsigned short* __restrict__ w1p) {
    int t = blockIdx.x * blockDim.x + threadIdx.x;
    if (t < 32768) {               // W0: 8 n_tiles x 8 k_tiles x 32 lanes x 16
        int j = t & 15, l = (t >> 4) & 31, frag = t >> 9;
        int kt = frag & 7, nt = frag >> 3;
        int n = (nt << 4) + (l & 15);
        int k = (kt << 5) + ((l & 16) ? 16 : 0) + j;
        w0p[t] = f2bf(W0[k * 128 + n]);
    } else if (t < 49152) {        // W1: 8 n_tiles x 4 k_tiles x 32 lanes x 16
        int e = t - 32768;
        int j = e & 15, l = (e >> 4) & 31, frag = e >> 9;
        int kt = frag & 3, nt = frag >> 2;
        int n = (nt << 4) + (l & 15);
        int k = (kt << 5) + ((l & 16) ? 16 : 0) + j;
        w1p[e] = f2bf(W1[k * 128 + n]);
    }
}

// ---------- main fused kernel ----------
// 256 threads = 8 waves; block handles 80 HR rows = 5 M-tiles of 16.
// Wave w owns cols [16w,16w+16); each B fragment is reused across 5 M-tiles.
#define MT 5
__global__ __launch_bounds__(256)
void mesh_main_kernel(const float* __restrict__ v,
                      const float* __restrict__ ehr,
                      const int*   __restrict__ idx,
                      const float* __restrict__ Wenc,
                      const float* __restrict__ benc,
                      const float* __restrict__ lng,
                      const float* __restrict__ lnb,
                      const float* __restrict__ b0,
                      const float* __restrict__ b1,
                      const unsigned short* __restrict__ w0p,
                      const unsigned short* __restrict__ w1p,
                      float* __restrict__ vlr,
                      float* __restrict__ counts) {
    __shared__ __align__(16) unsigned short hA[MT * 16 * 256]; // 40 KB, LN'd concat
    __shared__ __align__(16) unsigned short hB[MT * 16 * 128]; // 20 KB, layer-0 act
    __shared__ float sS[256];
    __shared__ float sQ[256];
    __shared__ int   sidx[MT * 16];

    const int tid     = threadIdx.x;
    const int rowbase = blockIdx.x * (MT * 16);

    if (tid < MT * 16) {
        int s = idx[rowbase + tid];
        sidx[tid] = s;
        atomicAdd(&counts[s], 1.0f);
    }

    // ----- build h = concat(e_enc, v), LayerNorm; 5 sub-passes of 16 rows -----
    const int r  = tid >> 4;    // row within sub-tile
    const int cg = tid & 15;    // 16-col group
    const int c0 = cg << 4;

    for (int p = 0; p < MT; ++p) {
        const int trow = p * 16 + r;           // row within block tile
        const int row  = rowbase + trow;

        float x[16];
        if (cg < 8) {                       // cols 0..127: e = s*W_enc + b_enc
            float es = ehr[row];
            #pragma unroll
            for (int j = 0; j < 16; ++j) x[j] = es * Wenc[c0 + j] + benc[c0 + j];
        } else {                            // cols 128..255: v
            const float4* vp = (const float4*)(v + (size_t)row * 128 + (c0 - 128));
            #pragma unroll
            for (int q = 0; q < 4; ++q) {
                float4 f = vp[q];
                x[4 * q + 0] = f.x; x[4 * q + 1] = f.y;
                x[4 * q + 2] = f.z; x[4 * q + 3] = f.w;
            }
        }
        float s = 0.f, q = 0.f;
        #pragma unroll
        for (int j = 0; j < 16; ++j) { s += x[j]; q += x[j] * x[j]; }
        sS[tid] = s; sQ[tid] = q;
        __syncthreads();

        float ts = 0.f, tq = 0.f;
        #pragma unroll
        for (int j = 0; j < 16; ++j) { ts += sS[(r << 4) + j]; tq += sQ[(r << 4) + j]; }
        const float mu  = ts * (1.f / 256.f);
        const float var = tq * (1.f / 256.f) - mu * mu;
        const float rs  = rsqrtf(var + 1e-5f);

        unsigned* hA32 = (unsigned*)hA;
        #pragma unroll
        for (int j = 0; j < 16; j += 2) {
            float y0 = (x[j]     - mu) * rs * lng[c0 + j]     + lnb[c0 + j];
            float y1 = (x[j + 1] - mu) * rs * lng[c0 + j + 1] + lnb[c0 + j + 1];
            hA32[(trow << 7) + ((c0 + j) >> 1)] =
                (unsigned)f2bf(y0) | ((unsigned)f2bf(y1) << 16);
        }
        __syncthreads();   // protect sS/sQ reuse + publish hA rows
    }

    // ----- layer 0: [80x256] x [256x128] via WMMA bf16; B reused across M-tiles -----
    const int lane  = tid & 31;
    const int w     = tid >> 5;                // wave id = n_tile
    const int n     = (w << 4) + (lane & 15);  // output column
    const int Mbase = (lane & 16) ? 8 : 0;

    v8f acc[MT];
    #pragma unroll
    for (int mt = 0; mt < MT; ++mt) acc[mt] = (v8f){0.f,0.f,0.f,0.f,0.f,0.f,0.f,0.f};

    #pragma unroll
    for (int kt = 0; kt < 8; ++kt) {
        v16bf b = load_b_frag(w0p, (w << 3) + kt, lane);
        #pragma unroll
        for (int mt = 0; mt < MT; ++mt) {
            v16bf a = load_a_frag(hA + mt * 16 * 256, lane, kt << 5, 256);
            acc[mt] = __builtin_amdgcn_wmma_f32_16x16x32_bf16(
                false, a, false, b, (short)0, acc[mt], false, false);
        }
    }
    const float bn0 = b0[n];
    #pragma unroll
    for (int mt = 0; mt < MT; ++mt) {
        #pragma unroll
        for (int rr = 0; rr < 8; ++rr) {
            float t = selu_f(acc[mt][rr] + bn0);
            hB[(mt * 16 + rr + Mbase) * 128 + n] = f2bf(t);
        }
    }
    __syncthreads();

    // ----- layer 1: [80x128] x [128x128] via WMMA bf16 -----
    #pragma unroll
    for (int mt = 0; mt < MT; ++mt) acc[mt] = (v8f){0.f,0.f,0.f,0.f,0.f,0.f,0.f,0.f};

    #pragma unroll
    for (int kt = 0; kt < 4; ++kt) {
        v16bf b = load_b_frag(w1p, (w << 2) + kt, lane);
        #pragma unroll
        for (int mt = 0; mt < MT; ++mt) {
            v16bf a = load_a_frag(hB + mt * 16 * 128, lane, kt << 5, 128);
            acc[mt] = __builtin_amdgcn_wmma_f32_16x16x32_bf16(
                false, a, false, b, (short)0, acc[mt], false, false);
        }
    }
    const float bn1 = b1[n];
    #pragma unroll
    for (int mt = 0; mt < MT; ++mt) {
        #pragma unroll
        for (int rr = 0; rr < 8; ++rr) {
            float t = selu_f(acc[mt][rr] + bn1);
            int M   = mt * 16 + rr + Mbase;
            atomicAdd(&vlr[(size_t)sidx[M] * 128 + n], t);
        }
    }
}

// ---------- scatter-mean finalize ----------
__global__ void finalize_kernel(float4* __restrict__ vlr4,
                                const float* __restrict__ counts, int n4) {
    int g = blockIdx.x * blockDim.x + threadIdx.x;
    if (g >= n4) return;
    int seg = g >> 5;                      // 32 float4 per 128-wide segment
    float inv = 1.f / fmaxf(counts[seg], 1.f);
    float4 f = vlr4[g];
    f.x *= inv; f.y *= inv; f.z *= inv; f.w *= inv;
    vlr4[g] = f;
}

// ---------- e_lr = edge_attr_lr @ W_elr + b_elr (K=3, streaming) ----------
__global__ __launch_bounds__(256)
void elr_kernel(const float* __restrict__ ea,
                const float* __restrict__ W,
                const float* __restrict__ b,
                float* __restrict__ out) {
    int gid = blockIdx.x * 256 + threadIdx.x;   // 32M threads total
    int row = gid >> 5;
    int c4  = (gid & 31) << 2;
    float a0 = ea[(size_t)row * 3 + 0];
    float a1 = ea[(size_t)row * 3 + 1];
    float a2 = ea[(size_t)row * 3 + 2];
    float4 w0 = *(const float4*)(W + c4);
    float4 w1 = *(const float4*)(W + 128 + c4);
    float4 w2 = *(const float4*)(W + 256 + c4);
    float4 bb = *(const float4*)(b + c4);
    f32x4 o;
    o.x = fmaf(a0, w0.x, fmaf(a1, w1.x, fmaf(a2, w2.x, bb.x)));
    o.y = fmaf(a0, w0.y, fmaf(a1, w1.y, fmaf(a2, w2.y, bb.y)));
    o.z = fmaf(a0, w0.z, fmaf(a1, w1.z, fmaf(a2, w2.z, bb.z)));
    o.w = fmaf(a0, w0.w, fmaf(a1, w1.w, fmaf(a2, w2.w, bb.w)));
    // write-once 512MB stream: non-temporal store keeps L2 for weights/atomics
    __builtin_nontemporal_store(o, (f32x4*)(out + (size_t)row * 128 + c4));
}

extern "C" void kernel_launch(void* const* d_in, const int* in_sizes, int n_in,
                              void* d_out, int out_size, void* d_ws, size_t ws_size,
                              hipStream_t stream) {
    const float* v        = (const float*)d_in[0];   // [500000,128]
    const float* ehr      = (const float*)d_in[1];   // [500000,1]
    const float* ea       = (const float*)d_in[2];   // [1000000,3]
    const int*   idx      = (const int*)  d_in[3];   // [500000]
    const float* Wenc     = (const float*)d_in[4];   // [1,128]
    const float* benc     = (const float*)d_in[5];   // [128]
    const float* lng      = (const float*)d_in[6];   // [256]
    const float* lnb      = (const float*)d_in[7];   // [256]
    const float* W0       = (const float*)d_in[8];   // [256,128]
    const float* b0       = (const float*)d_in[9];   // [128]
    const float* W1       = (const float*)d_in[10];  // [128,128]
    const float* b1       = (const float*)d_in[11];  // [128]
    const float* W_elr    = (const float*)d_in[12];  // [3,128]
    const float* b_elr    = (const float*)d_in[13];  // [128]

    float* out  = (float*)d_out;
    float* vlr  = out;                  // 125000*128 = 16,000,000 floats
    float* elr  = out + 16000000;       // 1000000*128 floats

    char* ws = (char*)d_ws;
    unsigned short* w0p    = (unsigned short*)(ws);           // 64 KB
    unsigned short* w1p    = (unsigned short*)(ws + 65536);   // 32 KB
    float*          counts = (float*)(ws + 98304);            // 500 KB

    // zero accumulators (d_out / d_ws are poisoned by the harness)
    zero_f4_kernel<<<(4000000 + 255) / 256, 256, 0, stream>>>((float4*)vlr, 4000000);
    zero_f4_kernel<<<(31250  + 255) / 256, 256, 0, stream>>>((float4*)counts, 31250);

    // repack weights to bf16 WMMA B-fragment order
    pack_weights_kernel<<<(49152 + 255) / 256, 256, 0, stream>>>(W0, W1, w0p, w1p);

    // fused encode + LN + 2x WMMA GEMM + SELU + scatter-add (500000/80 = 6250 blocks)
    mesh_main_kernel<<<6250, 256, 0, stream>>>(v, ehr, idx, Wenc, benc, lng, lnb,
                                               b0, b1, w0p, w1p, vlr, counts);

    // divide by counts -> mean
    finalize_kernel<<<(4000000 + 255) / 256, 256, 0, stream>>>((float4*)vlr, counts, 4000000);

    // independent streaming GEMM for e_lr (1e6*32/256 = 125000 blocks)
    elr_kernel<<<125000, 256, 0, stream>>>(ea, W_elr, b_elr, elr);
}
